// LstmAutoencoder_12730283065956
// MI455X (gfx1250) — matrix operand order repaired and checked by reading, not compiled, not run on hardware
//
#include <hip/hip_runtime.h>

// LSTM autoencoder (T=512, B=256, H=512) for MI455X / gfx1250.
//
// - Recurrent GEMMs as bf16 WMMA (v_wmma_f32_16x16x32_bf16), fp32 accum.
// - A-panel (h rows, shared by all 8 waves of a block) staged in LDS via
//   CDNA5 async copies (GLOBAL_LOAD_ASYNC_TO_LDS_B128 / s_wait_asynccnt),
//   with a synchronous fallback if the builtin is unavailable.
// - K-loop is software-pipelined (register double-buffered fragments) so
//   WMMAs overlap with the next tile's loads instead of s_wait_loadcnt 0.
// - Each wave owns a 16x16 (batch x hidden) tile and all 4 gates; the
//   sigmoid/tanh + c/h update is fused in-register.
// - h is double-buffered across steps; c is exclusively owned.
// - size-1 softmax == 1.0 exactly -> d_out filled with 1.0f (bit exact).

#define DEVINL __device__ __forceinline__

typedef __bf16 v16bf __attribute__((ext_vector_type(16)));
typedef __bf16 v8bf  __attribute__((ext_vector_type(8)));
typedef float  v8f   __attribute__((ext_vector_type(8)));
typedef int    v4i   __attribute__((ext_vector_type(4)));

#if __has_builtin(__builtin_amdgcn_global_load_async_to_lds_b128)
#define ASYNC_LDS 1
typedef __attribute__((address_space(1))) v4i* gv4i_p;  // global v4i*
typedef __attribute__((address_space(3))) v4i* lv4i_p;  // LDS v4i*
#if __has_builtin(__builtin_amdgcn_s_wait_asynccnt)
#define WAIT_ASYNC() __builtin_amdgcn_s_wait_asynccnt(0)
#else
#define WAIT_ASYNC() asm volatile("s_wait_asynccnt 0x0" ::: "memory")
#endif
#endif

namespace {
constexpr int kT    = 512;   // timesteps
constexpr int kB    = 256;   // batch
constexpr int kH    = 512;   // hidden
constexpr int kG4   = 4 * kH;
constexpr int kBH   = kB * kH;
constexpr int kPadH = kH + 8;  // LDS row stride (bf16): 1040 B -> 4-bank rotate
}

DEVINL __bf16 f2bf(float f) {
  union { float f; unsigned u; } a; a.f = f;
  unsigned u = a.u;
  unsigned r = u + 0x7FFFu + ((u >> 16) & 1u);   // round-to-nearest-even
  union { unsigned short s; __bf16 b; } o;
  o.s = (unsigned short)(r >> 16);
  return o.b;
}

DEVINL float sigmoidf(float x) { return 1.0f / (1.0f + __expf(-x)); }

// A-matrix 16x32 bf16 fragment from an LDS panel (padded row stride kPadH):
// lane holds row m = lane%16; halves e -> K = (lane/16)*16 + e.
DEVINL v16bf lds_a_frag(const __bf16* __restrict__ row, int kt) {
  const v8bf* p = reinterpret_cast<const v8bf*>(row + kt);
  v8bf lo = p[0];
  v8bf hi = p[1];                 // +8 elements
  v16bf r;
#pragma unroll
  for (int i = 0; i < 8; ++i) { r[i] = lo[i]; r[8 + i] = hi[i]; }
  return r;
}

// B-matrix 32x16 bf16 fragment: lane holds col n = lane%16;
// halves e -> K = (lane/16)*16 + e. B[k][n] = W[n][k] (row-major W).
DEVINL v16bf glb_b_frag(const __bf16* __restrict__ row, int kt) {
  const v8bf* p = reinterpret_cast<const v8bf*>(row + kt);
  v8bf lo = p[0];
  v8bf hi = p[1];
  v16bf r;
#pragma unroll
  for (int i = 0; i < 8; ++i) { r[i] = lo[i]; r[8 + i] = hi[i]; }
  return r;
}

DEVINL v8f wmma_bf16(v16bf a, v16bf b, v8f c) {
  return __builtin_amdgcn_wmma_f32_16x16x32_bf16(
      /*neg_a=*/false, a, /*neg_b=*/false, b,
      /*c_mod=*/(short)0, c, /*reuse_a=*/false, /*reuse_b=*/false);
}

// Stage a 16x512 bf16 panel (row-major, row stride kH) into padded LDS.
DEVINL void stage_panel(const __bf16* __restrict__ gsrc, __bf16* lds) {
  // 16 rows x 64 sixteen-byte chunks = 1024 chunks; 4 per thread.
#pragma unroll
  for (int it = 0; it < 4; ++it) {
    const int c   = it * 256 + threadIdx.x;
    const int row = c >> 6;
    const int col = (c & 63) * 8;          // bf16 elements
#ifdef ASYNC_LDS
    __builtin_amdgcn_global_load_async_to_lds_b128(
        (gv4i_p)(gsrc + row * kH + col),
        (lv4i_p)(lds + row * kPadH + col),
        0, 0);
#else
    *reinterpret_cast<v8bf*>(lds + row * kPadH + col) =
        *reinterpret_cast<const v8bf*>(gsrc + row * kH + col);
#endif
  }
}

// Software-pipelined K=512 panel GEMM: A from LDS, B (4 gate blocks) from
// global. Next tile's fragments are loaded before current WMMAs issue.
DEVINL void gemm_panel(const __bf16* __restrict__ ldsRow,
                       const __bf16* __restrict__ b0,
                       const __bf16* __restrict__ b1,
                       const __bf16* __restrict__ b2,
                       const __bf16* __restrict__ b3,
                       v8f& acc0, v8f& acc1, v8f& acc2, v8f& acc3) {
  v16bf aC  = lds_a_frag(ldsRow, 0);
  v16bf bC0 = glb_b_frag(b0, 0);
  v16bf bC1 = glb_b_frag(b1, 0);
  v16bf bC2 = glb_b_frag(b2, 0);
  v16bf bC3 = glb_b_frag(b3, 0);
  for (int kt = 32; kt < kH; kt += 32) {
    v16bf aN  = lds_a_frag(ldsRow, kt);
    v16bf bN0 = glb_b_frag(b0, kt);
    v16bf bN1 = glb_b_frag(b1, kt);
    v16bf bN2 = glb_b_frag(b2, kt);
    v16bf bN3 = glb_b_frag(b3, kt);
    acc0 = wmma_bf16(aC, bC0, acc0);
    acc1 = wmma_bf16(aC, bC1, acc1);
    acc2 = wmma_bf16(aC, bC2, acc2);
    acc3 = wmma_bf16(aC, bC3, acc3);
    aC = aN; bC0 = bN0; bC1 = bN1; bC2 = bN2; bC3 = bN3;
  }
  acc0 = wmma_bf16(aC, bC0, acc0);
  acc1 = wmma_bf16(aC, bC1, acc1);
  acc2 = wmma_bf16(aC, bC2, acc2);
  acc3 = wmma_bf16(aC, bC3, acc3);
}

// One fused LSTM cell step. 64 blocks x 256 threads = 512 waves; every wave
// of a block shares the same batch tile mt, each owns hidden tile jt.
//   wih  : [4H][H] bf16, layer-1 input weights (nullptr for layer 0)
//   whh  : [4H][H] bf16 recurrent weights
//   bias : [4H] f32, bih + bhh
//   xin  : [B][H] bf16 layer-1 input (= layer-0 h of same step), or nullptr
//   xsc  : [B] f32 scalar input x[t] (encoder layer 0), or nullptr
//   wih0 : [4H] f32 rank-1 input weights (encoder layer 0), or nullptr
//   hprev/hout : [B][H] bf16 (double buffered)
//   c    : [B][H] f32 (exclusive per wave element)
__global__ __launch_bounds__(256) void lstm_cell_kernel(
    const __bf16* __restrict__ wih,  const __bf16* __restrict__ whh,
    const float*  __restrict__ bias, const __bf16* __restrict__ xin,
    const float*  __restrict__ xsc,  const float*  __restrict__ wih0,
    const __bf16* __restrict__ hprev, __bf16* __restrict__ hout,
    float* __restrict__ c) {
  __shared__ __align__(16) __bf16 ldsA[2][16 * kPadH];

  const int lane = threadIdx.x & 31;
  const int wave = threadIdx.x >> 5;
  const int tile = blockIdx.x * 8 + wave;  // 0..511
  const int jt = tile & 31;                // hidden tile 0..31
  const int mt = tile >> 5;                // batch tile 0..15 (same for block)
  const int ln = lane & 15;
  const int lh = lane >> 4;

  // ---- stage shared A panels into LDS (async on CDNA5) ----
  stage_panel(hprev + (size_t)mt * 16 * kH, ldsA[0]);
  if (xin != nullptr) stage_panel(xin + (size_t)mt * 16 * kH, ldsA[1]);
#ifdef ASYNC_LDS
  WAIT_ASYNC();
#endif
  __syncthreads();

  v8f acc0 = {}, acc1 = {}, acc2 = {}, acc3 = {};  // gates i,f,g,o

  // Per-lane base pointers: A row in LDS, B rows (4 gates) in global.
  const __bf16* aRow = &ldsA[0][ln * kPadH + lh * 16];
  const int jrow = jt * 16 + ln;
  gemm_panel(aRow,
             whh + (size_t)(0 * kH + jrow) * kH + lh * 16,
             whh + (size_t)(1 * kH + jrow) * kH + lh * 16,
             whh + (size_t)(2 * kH + jrow) * kH + lh * 16,
             whh + (size_t)(3 * kH + jrow) * kH + lh * 16,
             acc0, acc1, acc2, acc3);
  if (xin != nullptr) {
    const __bf16* aRowX = &ldsA[1][ln * kPadH + lh * 16];
    gemm_panel(aRowX,
               wih + (size_t)(0 * kH + jrow) * kH + lh * 16,
               wih + (size_t)(1 * kH + jrow) * kH + lh * 16,
               wih + (size_t)(2 * kH + jrow) * kH + lh * 16,
               wih + (size_t)(3 * kH + jrow) * kH + lh * 16,
               acc0, acc1, acc2, acc3);
  }

  // Epilogue. C/D layout: element r of lane -> row m = (lane/16)*8 + r,
  // col n = lane%16.
  const int jb = jt * 16 + ln;
  const float bi = bias[jb];
  const float bf = bias[kH + jb];
  const float bg = bias[2 * kH + jb];
  const float bo = bias[3 * kH + jb];
  float w0 = 0.f, w1 = 0.f, w2 = 0.f, w3 = 0.f;
  if (xsc != nullptr) {            // rank-1 input term (scalar x per batch)
    w0 = wih0[jb]; w1 = wih0[kH + jb];
    w2 = wih0[2 * kH + jb]; w3 = wih0[3 * kH + jb];
  }
#pragma unroll
  for (int r = 0; r < 8; ++r) {
    const int bidx = mt * 16 + lh * 8 + r;
    float iv = acc0[r] + bi;
    float fv = acc1[r] + bf;
    float gv = acc2[r] + bg;
    float ov = acc3[r] + bo;
    if (xsc != nullptr) {
      const float xb = xsc[bidx];
      iv += xb * w0; fv += xb * w1; gv += xb * w2; ov += xb * w3;
    }
    iv = sigmoidf(iv);
    fv = sigmoidf(fv);
    gv = tanhf(gv);
    ov = sigmoidf(ov);
    const size_t idx = (size_t)bidx * kH + jb;
    const float cn = fv * c[idx] + iv * gv;
    c[idx] = cn;
    hout[idx] = f2bf(ov * tanhf(cn));
  }
}

__global__ void f32_to_bf16_kernel(const float* __restrict__ s,
                                   __bf16* __restrict__ d, int n) {
  int i = blockIdx.x * 256 + threadIdx.x;
  if (i < n) d[i] = f2bf(s[i]);
}

__global__ void bias_sum_kernel(const float* __restrict__ a,
                                const float* __restrict__ b,
                                float* __restrict__ o, int n) {
  int i = blockIdx.x * 256 + threadIdx.x;
  if (i < n) o[i] = a[i] + b[i];
}

__global__ void fill_u32_kernel(unsigned* __restrict__ p, unsigned v, int n) {
  int i = blockIdx.x * 256 + threadIdx.x;
  if (i < n) p[i] = v;
}

extern "C" void kernel_launch(void* const* d_in, const int* in_sizes, int n_in,
                              void* d_out, int out_size, void* d_ws,
                              size_t ws_size, hipStream_t stream) {
  (void)in_sizes; (void)n_in; (void)ws_size;
  const float* x         = (const float*)d_in[0];   // [T,B,1]
  const float* enc_Wih0  = (const float*)d_in[1];   // [4H,1]
  const float* enc_Whh0  = (const float*)d_in[2];   // [4H,H]
  const float* enc_bih0  = (const float*)d_in[3];
  const float* enc_bhh0  = (const float*)d_in[4];
  const float* enc_Wih1  = (const float*)d_in[5];
  const float* enc_Whh1  = (const float*)d_in[6];
  const float* enc_bih1  = (const float*)d_in[7];
  const float* enc_bhh1  = (const float*)d_in[8];
  // d_in[9] = dec_Wih0: decoder input is constant zero -> term vanishes.
  const float* dec_Whh0  = (const float*)d_in[10];
  const float* dec_bih0  = (const float*)d_in[11];
  const float* dec_bhh0  = (const float*)d_in[12];
  const float* dec_Wih1  = (const float*)d_in[13];
  const float* dec_Whh1  = (const float*)d_in[14];
  const float* dec_bih1  = (const float*)d_in[15];
  const float* dec_bhh1  = (const float*)d_in[16];
  // d_in[17]/[18] = lin_W/lin_b: size-1 softmax kills them (output == 1.0).

  char* base = (char*)d_ws;
  size_t off = 0;
  auto alloc = [&](size_t bytes) -> void* {
    void* p = base + off;
    off = (off + bytes + 255) & ~(size_t)255;
    return p;
  };
  const size_t WN = (size_t)kG4 * kH;  // elements per weight matrix
  __bf16* wEnc0  = (__bf16*)alloc(WN * 2);
  __bf16* wEncI1 = (__bf16*)alloc(WN * 2);
  __bf16* wEncH1 = (__bf16*)alloc(WN * 2);
  __bf16* wDec0  = (__bf16*)alloc(WN * 2);
  __bf16* wDecI1 = (__bf16*)alloc(WN * 2);
  __bf16* wDecH1 = (__bf16*)alloc(WN * 2);
  float*  bEnc0  = (float*)alloc(kG4 * 4);
  float*  bEnc1  = (float*)alloc(kG4 * 4);
  float*  bDec0  = (float*)alloc(kG4 * 4);
  float*  bDec1  = (float*)alloc(kG4 * 4);
  __bf16* hA[2]  = {(__bf16*)alloc(kBH * 2), (__bf16*)alloc(kBH * 2)};
  __bf16* hB[2]  = {(__bf16*)alloc(kBH * 2), (__bf16*)alloc(kBH * 2)};
  float*  cA     = (float*)alloc(kBH * 4);
  float*  cB     = (float*)alloc(kBH * 4);

  // ---- prep: bf16 weights, combined biases, zeroed initial state ----
  const int wBlocks = (int)((WN + 255) / 256);
  f32_to_bf16_kernel<<<wBlocks, 256, 0, stream>>>(enc_Whh0, wEnc0,  (int)WN);
  f32_to_bf16_kernel<<<wBlocks, 256, 0, stream>>>(enc_Wih1, wEncI1, (int)WN);
  f32_to_bf16_kernel<<<wBlocks, 256, 0, stream>>>(enc_Whh1, wEncH1, (int)WN);
  f32_to_bf16_kernel<<<wBlocks, 256, 0, stream>>>(dec_Whh0, wDec0,  (int)WN);
  f32_to_bf16_kernel<<<wBlocks, 256, 0, stream>>>(dec_Wih1, wDecI1, (int)WN);
  f32_to_bf16_kernel<<<wBlocks, 256, 0, stream>>>(dec_Whh1, wDecH1, (int)WN);
  bias_sum_kernel<<<kG4 / 256, 256, 0, stream>>>(enc_bih0, enc_bhh0, bEnc0, kG4);
  bias_sum_kernel<<<kG4 / 256, 256, 0, stream>>>(enc_bih1, enc_bhh1, bEnc1, kG4);
  bias_sum_kernel<<<kG4 / 256, 256, 0, stream>>>(dec_bih0, dec_bhh0, bDec0, kG4);
  bias_sum_kernel<<<kG4 / 256, 256, 0, stream>>>(dec_bih1, dec_bhh1, bDec1, kG4);
  fill_u32_kernel<<<kBH / 2 / 256, 256, 0, stream>>>((unsigned*)hA[0], 0u, kBH / 2);
  fill_u32_kernel<<<kBH / 2 / 256, 256, 0, stream>>>((unsigned*)hB[0], 0u, kBH / 2);
  fill_u32_kernel<<<kBH / 256, 256, 0, stream>>>((unsigned*)cA, 0u, kBH);
  fill_u32_kernel<<<kBH / 256, 256, 0, stream>>>((unsigned*)cB, 0u, kBH);

  // ---- encoder: 2-layer LSTM over T steps ----
  int pA = 0, pB = 0;
  for (int t = 0; t < kT; ++t) {
    lstm_cell_kernel<<<64, 256, 0, stream>>>(
        nullptr, wEnc0, bEnc0, nullptr, x + (size_t)t * kB, enc_Wih0,
        hA[pA], hA[1 - pA], cA);
    lstm_cell_kernel<<<64, 256, 0, stream>>>(
        wEncI1, wEncH1, bEnc1, hA[1 - pA], nullptr, nullptr,
        hB[pB], hB[1 - pB], cB);
    pA ^= 1; pB ^= 1;
  }

  // ---- decoder: carry = encoder final states; layer-0 input == 0 ----
  for (int t = 0; t < kT; ++t) {
    lstm_cell_kernel<<<64, 256, 0, stream>>>(
        nullptr, wDec0, bDec0, nullptr, nullptr, nullptr,
        hA[pA], hA[1 - pA], cA);
    lstm_cell_kernel<<<64, 256, 0, stream>>>(
        wDecI1, wDecH1, bDec1, hA[1 - pA], nullptr, nullptr,
        hB[pB], hB[1 - pB], cB);
    pA ^= 1; pB ^= 1;
  }

  // ---- output: softmax over a size-1 axis is exactly 1.0 ----
  fill_u32_kernel<<<(out_size + 255) / 256, 256, 0, stream>>>(
      (unsigned*)d_out, 0x3F800000u, out_size);
}
